// MaskedAutoencoderViT_62878321214029
// MI455X (gfx1250) — compile-verified, matrix operand order checked by reading, not run on hardware
//
#include <hip/hip_runtime.h>

// ---------------------------------------------------------------------------
// MI455X (gfx1250, wave32) point-cloud transformer.
//  * all matmuls: v_wmma_f32_16x16x32_bf16 (f32 accumulate)
//  * weights converted+transposed to bf16 once per layer; activations kept
//    bf16 between ops -> GEMM/attention staging is raw 16B copies
//  * staging uses GLOBAL_LOAD_ASYNC_TO_LDS_B128 (ASYNCcnt, double-buffered)
//    when the builtin exists; falls back to load+ds_store otherwise
// ---------------------------------------------------------------------------

typedef __bf16 bf16;
typedef __attribute__((ext_vector_type(16))) __bf16 v16bf;
typedef __attribute__((ext_vector_type(8)))  __bf16 v8bf;
typedef __attribute__((ext_vector_type(4)))  __bf16 v4bf;
typedef __attribute__((ext_vector_type(8)))  float  v8f;

#if defined(__gfx1250__) && __has_builtin(__builtin_amdgcn_global_load_async_to_lds_b128)
#define HAVE_ASYNC 1
#else
#define HAVE_ASYNC 0
#endif

#if HAVE_ASYNC
// builtin signature (from clang diagnostic): param0 = int4 addrspace(1)*,
// i.e. global source first, then LDS dest, imm offset, cpol.
typedef int g4i __attribute__((vector_size(16)));
typedef __attribute__((address_space(1))) g4i* gp16;
typedef __attribute__((address_space(3))) g4i* lp16;
#endif

__device__ __forceinline__ bf16 f2bf(float f) {
    union { float f; unsigned u; } a; a.f = f;
    unsigned r = a.u + 0x7FFFu + ((a.u >> 16) & 1u);   // round-to-nearest-even
    union { unsigned short s; bf16 b; } o; o.s = (unsigned short)(r >> 16);
    return o.b;
}

__device__ __forceinline__ v8f zero8() {
    v8f z = {0.f,0.f,0.f,0.f,0.f,0.f,0.f,0.f};
    return z;
}

// 16-byte global -> LDS copy; async (ASYNCcnt) when available.
__device__ __forceinline__ void cp16(void* lds, const void* g) {
#if HAVE_ASYNC
    __builtin_amdgcn_global_load_async_to_lds_b128((gp16)g, (lp16)lds, 0, 0);
#else
    *(uint4*)lds = *(const uint4*)g;
#endif
}

template <int N>
__device__ __forceinline__ void wait_async() {
#if HAVE_ASYNC
#if __has_builtin(__builtin_amdgcn_s_wait_asynccnt)
    __builtin_amdgcn_s_wait_asynccnt(N);
#else
    asm volatile("s_wait_asynccnt %0" :: "i"(N) : "memory");
#endif
#endif
}

// Load one 16-lane-striped A/B fragment (16x32 / 32x16 bf16) from an LDS row.
// ISA 7.12.2: lanes 0-15 take K chunks {0..7,16..23}, lanes 16-31 {8..15,24..31}.
__device__ __forceinline__ v16bf ld_frag(const bf16* base, int lane) {
    const int k0 = (lane & 16) ? 8 : 0;
    union { struct { v8bf lo; v8bf hi; } s; v16bf v; } u;
    u.s.lo = *(const v8bf*)(base + k0);
    u.s.hi = *(const v8bf*)(base + k0 + 16);
    return u.v;
}

#define WMMA_BF16(A, B, C) \
    __builtin_amdgcn_wmma_f32_16x16x32_bf16(false, (A), false, (B), (short)0, (C), false, false)

#define LSTR 40   // padded LDS row stride (bf16): 80B = 16B-aligned, bank-spread

// ---------------------------------------------------------------------------
// Weight convert + transpose: W[K][N] f32 -> Wt[N][K] bf16 (32x32 LDS tiles)
// ---------------------------------------------------------------------------
__global__ __launch_bounds__(256) void k_cvt_w(
    const float* __restrict__ W, bf16* __restrict__ Wt, int K, int N)
{
    __shared__ bf16 t[32 * 33];
    const int nn = blockIdx.x * 32;
    const int kk = blockIdx.y * 32;
    const int col = threadIdx.x & 31;
    const int r0  = threadIdx.x >> 5;
#pragma unroll
    for (int i = 0; i < 4; ++i) {
        const int row = r0 + i * 8;                       // k-local
        t[col * 33 + row] = f2bf(W[(size_t)(kk + row) * N + nn + col]);
    }
    __syncthreads();
#pragma unroll
    for (int i = 0; i < 4; ++i) {
        const int row = r0 + i * 8;                       // n-local
        Wt[(size_t)(nn + row) * K + kk + col] = t[row * 33 + col];
    }
}

// ---------------------------------------------------------------------------
// GEMM: C[M,N] = act(A[M,K] @ Wt[N,K]^T + bias) (+res). A,Wt bf16; C f32 or bf16.
// flags: 1=exact GELU, 2=+residual(f32), 4=bf16 output.
// 128x128 block tile, 8 waves x (32x64). Async double-buffered LDS staging.
// ---------------------------------------------------------------------------
__device__ __forceinline__ void stage_tiles(
    const bf16* __restrict__ A, const bf16* __restrict__ Wt,
    bf16* sA, bf16* sB, int m0, int n0, int k0, int M, int K, int tid)
{
#pragma unroll
    for (int i = 0; i < 2; ++i) {                         // A: 128 x 32
        const int id  = tid + i * 256;                    // 512 16B chunks
        const int row = id >> 2;
        const int c8  = (id & 3) * 8;
        int gr = m0 + row; if (gr >= M) gr = M - 1;
        cp16(sA + row * LSTR + c8, A + (size_t)gr * K + k0 + c8);
    }
#pragma unroll
    for (int i = 0; i < 2; ++i) {                         // Wt: 128 x 32
        const int id  = tid + i * 256;
        const int row = id >> 2;
        const int c8  = (id & 3) * 8;
        cp16(sB + row * LSTR + c8, Wt + (size_t)(n0 + row) * K + k0 + c8);
    }
}

__device__ __forceinline__ void mma_step(
    const bf16* sA, const bf16* sB, int wm, int wn, int lane, v8f (&acc)[2][4])
{
    v16bf af[2], bfm[4];
#pragma unroll
    for (int mi = 0; mi < 2; ++mi)
        af[mi] = ld_frag(sA + (wm * 32 + mi * 16 + (lane & 15)) * LSTR, lane);
#pragma unroll
    for (int ni = 0; ni < 4; ++ni)
        bfm[ni] = ld_frag(sB + (wn * 64 + ni * 16 + (lane & 15)) * LSTR, lane);
#pragma unroll
    for (int mi = 0; mi < 2; ++mi)
#pragma unroll
        for (int ni = 0; ni < 4; ++ni)
            acc[mi][ni] = WMMA_BF16(af[mi], bfm[ni], acc[mi][ni]);
}

__global__ __launch_bounds__(256) void k_gemm(
    const bf16* __restrict__ A, const bf16* __restrict__ Wt,
    const float* __restrict__ bias, const float* __restrict__ res,
    float* __restrict__ Cf, bf16* __restrict__ Cb,
    int M, int N, int K, int flags)
{
    __shared__ __align__(16) bf16 sA[2][128 * LSTR];
    __shared__ __align__(16) bf16 sB[2][128 * LSTR];

    const int tid  = threadIdx.x;
    const int lane = tid & 31;
    const int wave = tid >> 5;
    const int wm   = wave >> 1;             // 0..3 : 32 rows
    const int wn   = wave & 1;              // 0..1 : 64 cols
    const int n0   = blockIdx.x * 128;
    const int m0   = blockIdx.y * 128;

    v8f acc[2][4];
#pragma unroll
    for (int i = 0; i < 2; ++i)
#pragma unroll
        for (int j = 0; j < 4; ++j) acc[i][j] = zero8();

    const int nk = K >> 5;
#if HAVE_ASYNC
    // double-buffered: overlap async fills with WMMA; per-wave self-wait +
    // workgroup barrier gives cross-wave completion of the current buffer.
    stage_tiles(A, Wt, sA[0], sB[0], m0, n0, 0, M, K, tid);
    for (int kt = 0; kt < nk; ++kt) {
        const int cur = kt & 1;
        if (kt + 1 < nk)
            stage_tiles(A, Wt, sA[cur ^ 1], sB[cur ^ 1], m0, n0, (kt + 1) << 5, M, K, tid);
        if (kt + 1 < nk) wait_async<4>(); else wait_async<0>();
        __syncthreads();
        mma_step(sA[cur], sB[cur], wm, wn, lane, acc);
        __syncthreads();
    }
#else
    for (int kt = 0; kt < nk; ++kt) {
        stage_tiles(A, Wt, sA[0], sB[0], m0, n0, kt << 5, M, K, tid);
        __syncthreads();
        mma_step(sA[0], sB[0], wm, wn, lane, acc);
        __syncthreads();
    }
#endif

    const bool gelu   = flags & 1;
    const bool hasres = flags & 2;
    const bool outbf  = flags & 4;
    const int  rofs   = (lane & 16) ? 8 : 0;
#pragma unroll
    for (int mi = 0; mi < 2; ++mi) {
#pragma unroll
        for (int ni = 0; ni < 4; ++ni) {
            const int gn = n0 + wn * 64 + ni * 16 + (lane & 15);
#pragma unroll
            for (int r = 0; r < 8; ++r) {
                const int gm = m0 + wm * 32 + mi * 16 + r + rofs;
                if (gm < M) {
                    float v = acc[mi][ni][r] + bias[gn];
                    if (gelu) v = 0.5f * v * (1.0f + erff(v * 0.70710678118f));
                    if (hasres) v += res[(size_t)gm * N + gn];
                    if (outbf) Cb[(size_t)gm * N + gn] = f2bf(v);
                    else       Cf[(size_t)gm * N + gn] = v;
                }
            }
        }
    }
}

// ---------------------------------------------------------------------------
// Flash attention on bf16 qkv: grid (qblocks, HEADS, B), 128 thr = 4 waves,
// each wave owns 16 q rows, streams 32-key blocks. Dh=64, scale=1/8.
// qkv row layout: [b*T + t][3*1024] bf16, head offset h*64. Output bf16.
// ---------------------------------------------------------------------------
__global__ __launch_bounds__(128) void k_attn(
    const bf16* __restrict__ qkv, bf16* __restrict__ out, int T, int causal)
{
    const int b     = blockIdx.z;
    const int h     = blockIdx.y;
    const int qbase = blockIdx.x * 64;
    const int tid   = threadIdx.x;
    const int lane  = tid & 31;
    const int wave  = tid >> 5;
    const int rofs  = (lane & 16) ? 8 : 0;

    __shared__ __align__(16) bf16 sQ [64 * 72];
    __shared__ __align__(16) bf16 sK [32 * 72];
    __shared__ __align__(16) bf16 sVt[64 * LSTR];
    __shared__ __align__(16) bf16 sP [4][16 * LSTR];

    const size_t base = ((size_t)b * T) * 3072 + (size_t)h * 64;

    // stage Q block (64 x 64 bf16) -- async raw copies
#pragma unroll
    for (int i = 0; i < 4; ++i) {
        const int id  = tid + i * 128;                    // 512 16B chunks
        const int row = id >> 3;
        const int c8  = (id & 7) * 8;
        int gq = qbase + row; if (gq >= T) gq = T - 1;
        cp16(sQ + row * 72 + c8, qkv + base + (size_t)gq * 3072 + c8);
    }
    wait_async<0>();
    __syncthreads();

    const int qrow = wave * 16 + (lane & 15);
    v16bf qf[2];
    qf[0] = ld_frag(sQ + qrow * 72 +  0, lane);
    qf[1] = ld_frag(sQ + qrow * 72 + 32, lane);

    v8f o[4];
#pragma unroll
    for (int d = 0; d < 4; ++d) o[d] = zero8();
    float mrow[8], lrow[8];
#pragma unroll
    for (int r = 0; r < 8; ++r) { mrow[r] = -3e38f; lrow[r] = 0.f; }

    int nkb = (T + 31) >> 5;
    if (causal) { int ce = (qbase >> 5) + 2; if (ce < nkb) nkb = ce; }

    for (int kb = 0; kb < nkb; ++kb) {
        // K (32x64): async copies; Vt (64x32): transpose via ds_store_b16
#pragma unroll
        for (int i = 0; i < 2; ++i) {
            const int id  = tid + i * 128;                // 256 16B chunks
            const int row = id >> 3;
            const int c8  = (id & 7) * 8;
            int key = kb * 32 + row; if (key >= T) key = T - 1;
            cp16(sK + row * 72 + c8, qkv + base + 1024 + (size_t)key * 3072 + c8);
        }
#pragma unroll
        for (int i = 0; i < 4; ++i) {
            const int id  = tid + i * 128;                // 512 4-elem chunks
            const int row = id >> 4;
            const int c4  = (id & 15) * 4;
            int key = kb * 32 + row; if (key >= T) key = T - 1;
            const v4bf vv = *(const v4bf*)(qkv + base + 2048 + (size_t)key * 3072 + c4);
            sVt[(c4 + 0) * LSTR + row] = vv[0];
            sVt[(c4 + 1) * LSTR + row] = vv[1];
            sVt[(c4 + 2) * LSTR + row] = vv[2];
            sVt[(c4 + 3) * LSTR + row] = vv[3];
        }
        wait_async<0>();
        __syncthreads();

        // ---- S = Q K^T over 32 keys (two 16-key column tiles) ----
        v8f s0 = zero8(), s1 = zero8();
#pragma unroll
        for (int c = 0; c < 2; ++c) {
            v16bf bk0 = ld_frag(sK + ((lane & 15))      * 72 + c * 32, lane);
            v16bf bk1 = ld_frag(sK + (16 + (lane & 15)) * 72 + c * 32, lane);
            s0 = WMMA_BF16(qf[c], bk0, s0);
            s1 = WMMA_BF16(qf[c], bk1, s1);
        }

        // ---- online softmax (row reductions within 16-lane halves) ----
        const float scale = 0.125f;
        const int   col0  = kb * 32 + (lane & 15);
        const int   col1  = col0 + 16;
        float p0[8], p1[8], alpha[8];
#pragma unroll
        for (int r = 0; r < 8; ++r) {
            const int gq = qbase + wave * 16 + r + rofs;
            float v0 = s0[r] * scale, v1 = s1[r] * scale;
            if (col0 >= T || gq >= T || (causal && col0 > gq)) v0 = -1e9f;
            if (col1 >= T || gq >= T || (causal && col1 > gq)) v1 = -1e9f;
            float mx = fmaxf(v0, v1);
#pragma unroll
            for (int m = 8; m >= 1; m >>= 1) mx = fmaxf(mx, __shfl_xor(mx, m, 32));
            const float mnew = fmaxf(mrow[r], mx);
            alpha[r] = __expf(mrow[r] - mnew);
            p0[r] = __expf(v0 - mnew);
            p1[r] = __expf(v1 - mnew);
            float rs = p0[r] + p1[r];
#pragma unroll
            for (int m = 8; m >= 1; m >>= 1) rs += __shfl_xor(rs, m, 32);
            lrow[r] = lrow[r] * alpha[r] + rs;
            mrow[r] = mnew;
        }
#pragma unroll
        for (int d = 0; d < 4; ++d)
#pragma unroll
            for (int r = 0; r < 8; ++r) o[d][r] *= alpha[r];

        // ---- P through LDS (C-layout -> A-fragment layout), O += P V ----
        bf16* sp = sP[wave];
#pragma unroll
        for (int r = 0; r < 8; ++r) {
            sp[(r + rofs) * LSTR + (lane & 15)]      = f2bf(p0[r]);
            sp[(r + rofs) * LSTR + 16 + (lane & 15)] = f2bf(p1[r]);
        }
        const v16bf pf = ld_frag(sp + (lane & 15) * LSTR, lane);
#pragma unroll
        for (int d = 0; d < 4; ++d) {
            v16bf vf = ld_frag(sVt + (d * 16 + (lane & 15)) * LSTR, lane);
            o[d] = WMMA_BF16(pf, vf, o[d]);
        }
        __syncthreads();
    }

    // ---- write O / l as bf16 to out[b,t, h*64 + dh] ----
#pragma unroll
    for (int d = 0; d < 4; ++d)
#pragma unroll
        for (int r = 0; r < 8; ++r) {
            const int gq = qbase + wave * 16 + r + rofs;
            if (gq < T)
                out[((size_t)b * T + gq) * 1024 + h * 64 + d * 16 + (lane & 15)]
                    = f2bf(o[d][r] / lrow[r]);
        }
}

// ---------------------------------------------------------------------------
// LayerNorm over D=1024, one token per 256-thread block. f32 and bf16 outs.
// ---------------------------------------------------------------------------
__device__ __forceinline__ float4 ln_core(
    const float* __restrict__ x, const float* __restrict__ g,
    const float* __restrict__ bb, int tok, int d)
{
    const float4 v = *(const float4*)(x + (size_t)tok * 1024 + d);
    float s  = v.x + v.y + v.z + v.w;
    float s2 = v.x*v.x + v.y*v.y + v.z*v.z + v.w*v.w;
#pragma unroll
    for (int m = 16; m >= 1; m >>= 1) {
        s  += __shfl_xor(s,  m, 32);
        s2 += __shfl_xor(s2, m, 32);
    }
    __shared__ float red[16];
    const int wave = threadIdx.x >> 5, lane = threadIdx.x & 31;
    if (lane == 0) { red[wave] = s; red[8 + wave] = s2; }
    __syncthreads();
    float ts = 0.f, ts2 = 0.f;
#pragma unroll
    for (int i = 0; i < 8; ++i) { ts += red[i]; ts2 += red[8 + i]; }
    const float mu  = ts * (1.0f / 1024.0f);
    const float var = ts2 * (1.0f / 1024.0f) - mu * mu;
    const float rs  = rsqrtf(var + 1e-5f);
    const float4 gg = *(const float4*)(g + d);
    const float4 bv = *(const float4*)(bb + d);
    float4 o;
    o.x = (v.x - mu) * rs * gg.x + bv.x;
    o.y = (v.y - mu) * rs * gg.y + bv.y;
    o.z = (v.z - mu) * rs * gg.z + bv.z;
    o.w = (v.w - mu) * rs * gg.w + bv.w;
    return o;
}

__global__ __launch_bounds__(256) void k_ln(
    const float* __restrict__ x, const float* __restrict__ g,
    const float* __restrict__ bb, float* __restrict__ y)
{
    const int tok = blockIdx.x, d = threadIdx.x * 4;
    const float4 o = ln_core(x, g, bb, tok, d);
    *(float4*)(y + (size_t)tok * 1024 + d) = o;
}

__global__ __launch_bounds__(256) void k_ln_bf(
    const float* __restrict__ x, const float* __restrict__ g,
    const float* __restrict__ bb, bf16* __restrict__ y)
{
    const int tok = blockIdx.x, d = threadIdx.x * 4;
    const float4 o = ln_core(x, g, bb, tok, d);
    v4bf b4; b4[0] = f2bf(o.x); b4[1] = f2bf(o.y);
             b4[2] = f2bf(o.z); b4[3] = f2bf(o.w);
    *(v4bf*)(y + (size_t)tok * 1024 + d) = b4;
}

// D->1 projection per token (pred head & final head).
__global__ __launch_bounds__(256) void k_rowdot(
    const float* __restrict__ x, const float* __restrict__ w,
    const float* __restrict__ b0, float* __restrict__ out)
{
    const int tok = blockIdx.x;
    const int d   = threadIdx.x * 4;
    const float4 v  = *(const float4*)(x + (size_t)tok * 1024 + d);
    const float4 wv = *(const float4*)(w + d);
    float s = v.x*wv.x + v.y*wv.y + v.z*wv.z + v.w*wv.w;
#pragma unroll
    for (int m = 16; m >= 1; m >>= 1) s += __shfl_xor(s, m, 32);
    __shared__ float red[8];
    const int wave = threadIdx.x >> 5, lane = threadIdx.x & 31;
    if (lane == 0) red[wave] = s;
    __syncthreads();
    if (threadIdx.x == 0) {
        float t = 0.f;
#pragma unroll
        for (int i = 0; i < 8; ++i) t += red[i];
        out[tok] = t + b0[0];
    }
}

// Exact jax.lax.top_k ordering via rank counting (desc value, asc index).
// Softmax is monotone so ranking raw logits is exact.
__global__ void k_topk(const float* __restrict__ logits, int* __restrict__ sel)
{
    const int b = blockIdx.x;
    const float* l = logits + b * 2048;
    for (int i = threadIdx.x; i < 2048; i += blockDim.x) {
        const float li = l[i];
        int rank = 0;
        for (int j = 0; j < 2048; ++j) {
            const float lj = l[j];
            rank += (lj > li) || (lj == li && j < i);
        }
        if (rank < 1024) sel[b * 1024 + rank] = i;
    }
}

// h[tok][d] = xyz . pos_w + pos_b
__global__ void k_embed_pc(const float* __restrict__ x, const float* __restrict__ w,
                           const float* __restrict__ bias, float* __restrict__ h)
{
    const int tok = blockIdx.x;
    const int d   = blockIdx.y * 256 + threadIdx.x;
    const float* xr = x + (size_t)tok * 4;
    h[(size_t)tok * 1024 + d] =
        xr[0] * w[d] + xr[1] * w[1024 + d] + xr[2] * w[2048 + d] + bias[d];
}

// AR token embed from gathered points; r in [0,1023) (drops last point).
__global__ void k_embed_ar(const float* __restrict__ x, const int* __restrict__ sel,
                           const float* __restrict__ pw, const float* __restrict__ pb,
                           const float* __restrict__ vw, const float* __restrict__ vb,
                           float* __restrict__ tok)
{
    const int b = blockIdx.z;
    const int r = blockIdx.x;
    const int d = blockIdx.y * 256 + threadIdx.x;
    const int src = sel[b * 1024 + r];
    const float* xr = x + ((size_t)b * 2048 + src) * 4;
    tok[((size_t)b * 1023 + r) * 1024 + d] =
        xr[0] * pw[d] + xr[1] * pw[1024 + d] + xr[2] * pw[2048 + d] + pb[d]
        + xr[3] * vw[d] + vb[d];
}

// second output: points[:,1:,3:]  ->  2046 scalars
__global__ void k_out2(const float* __restrict__ x, const int* __restrict__ sel,
                       float* __restrict__ o2)
{
    const int i = blockIdx.x * blockDim.x + threadIdx.x;
    if (i >= 2046) return;
    const int b = i / 1023, r = i % 1023;
    const int src = sel[b * 1024 + r + 1];
    o2[i] = x[((size_t)b * 2048 + src) * 4 + 3];
}

// ---------------------------------------------------------------------------
// Host orchestration. Input order = jax pytree (alphabetical dict keys).
// ---------------------------------------------------------------------------
extern "C" void kernel_launch(void* const* d_in, const int* in_sizes, int n_in,
                              void* d_out, int out_size, void* d_ws, size_t ws_size,
                              hipStream_t stream)
{
    (void)in_sizes; (void)n_in; (void)out_size; (void)ws_size;
    const float* ar_b1    = (const float*)d_in[0];
    const float* ar_b2    = (const float*)d_in[1];
    const float* ar_bo    = (const float*)d_in[2];
    const float* ar_bqkv  = (const float*)d_in[3];
    const float* ar_ln1b  = (const float*)d_in[4];
    const float* ar_ln1g  = (const float*)d_in[5];
    const float* ar_ln2b  = (const float*)d_in[6];
    const float* ar_ln2g  = (const float*)d_in[7];
    const float* ar_w1    = (const float*)d_in[8];
    const float* ar_w2    = (const float*)d_in[9];
    const float* ar_wo    = (const float*)d_in[10];
    const float* ar_wqkv  = (const float*)d_in[11];
    const float* ar_nb    = (const float*)d_in[12];
    const float* ar_ng    = (const float*)d_in[13];
    const float* ar_pos_b = (const float*)d_in[14];
    const float* ar_pos_w = (const float*)d_in[15];
    const float* ar_val_b = (const float*)d_in[16];
    const float* ar_val_w = (const float*)d_in[17];
    const float* head_b   = (const float*)d_in[18];
    const float* head_w   = (const float*)d_in[19];
    const float* pc_b1    = (const float*)d_in[20];
    const float* pc_b2    = (const float*)d_in[21];
    const float* pc_bo    = (const float*)d_in[22];
    const float* pc_bqkv  = (const float*)d_in[23];
    const float* pc_ln1b  = (const float*)d_in[24];
    const float* pc_ln1g  = (const float*)d_in[25];
    const float* pc_ln2b  = (const float*)d_in[26];
    const float* pc_ln2g  = (const float*)d_in[27];
    const float* pc_w1    = (const float*)d_in[28];
    const float* pc_w2    = (const float*)d_in[29];
    const float* pc_wo    = (const float*)d_in[30];
    const float* pc_wqkv  = (const float*)d_in[31];
    const float* pc_nb    = (const float*)d_in[32];
    const float* pc_ng    = (const float*)d_in[33];
    const float* pos_b    = (const float*)d_in[34];
    const float* pos_w    = (const float*)d_in[35];
    const float* pred_b   = (const float*)d_in[36];
    const float* pred_w   = (const float*)d_in[37];
    const float* x        = (const float*)d_in[38];

    // Workspace (~104 MB): h | tmp_bf | qkv_bf | ff_bf | wt | tmpf | logits | sel
    char*  w8     = (char*)d_ws;
    float* h      = (float*)(w8);                               // 16 MB
    bf16*  tmp_bf = (bf16*) (w8 + ((size_t)16  << 20));         //  8 MB
    bf16*  qkv_bf = (bf16*) (w8 + ((size_t)24  << 20));         // 24 MB
    bf16*  ff_bf  = (bf16*) (w8 + ((size_t)48  << 20));         // 32 MB
    bf16*  wt     = (bf16*) (w8 + ((size_t)80  << 20));         //  8 MB
    float* tmpf   = (float*)(w8 + ((size_t)88  << 20));         // 16 MB
    float* logits = (float*)(w8 + ((size_t)104 << 20));
    int*   sel    = (int*)  (w8 + ((size_t)104 << 20) + 32768);

    auto run_stack = [&](float* hbuf, int M, int T, int L, int causal,
                         const float* b1, const float* b2, const float* bo,
                         const float* bqkv, const float* ln1b, const float* ln1g,
                         const float* ln2b, const float* ln2g, const float* w1,
                         const float* w2, const float* wo, const float* wqkv) {
        const int gM = (M + 127) / 128;
        const int qb = (T + 63) / 64;
        for (int l = 0; l < L; ++l) {
            k_ln_bf<<<M, 256, 0, stream>>>(hbuf, ln1g + (size_t)l*1024, ln1b + (size_t)l*1024, tmp_bf);
            k_cvt_w<<<dim3(96, 32), 256, 0, stream>>>(wqkv + (size_t)l*1024*3072, wt, 1024, 3072);
            k_gemm<<<dim3(24, gM), 256, 0, stream>>>(tmp_bf, wt, bqkv + (size_t)l*3072,
                nullptr, nullptr, qkv_bf, M, 3072, 1024, 4);
            k_attn<<<dim3(qb, 16, 2), 128, 0, stream>>>(qkv_bf, tmp_bf, T, causal);
            k_cvt_w<<<dim3(32, 32), 256, 0, stream>>>(wo + (size_t)l*1024*1024, wt, 1024, 1024);
            k_gemm<<<dim3(8, gM), 256, 0, stream>>>(tmp_bf, wt, bo + (size_t)l*1024,
                hbuf, hbuf, nullptr, M, 1024, 1024, 2);
            k_ln_bf<<<M, 256, 0, stream>>>(hbuf, ln2g + (size_t)l*1024, ln2b + (size_t)l*1024, tmp_bf);
            k_cvt_w<<<dim3(128, 32), 256, 0, stream>>>(w1 + (size_t)l*1024*4096, wt, 1024, 4096);
            k_gemm<<<dim3(32, gM), 256, 0, stream>>>(tmp_bf, wt, b1 + (size_t)l*4096,
                nullptr, nullptr, ff_bf, M, 4096, 1024, 5);        // GELU + bf16 out
            k_cvt_w<<<dim3(32, 128), 256, 0, stream>>>(w2 + (size_t)l*4096*1024, wt, 4096, 1024);
            k_gemm<<<dim3(8, gM), 256, 0, stream>>>(ff_bf, wt, b2 + (size_t)l*1024,
                hbuf, hbuf, nullptr, M, 1024, 4096, 2);
        }
    };

    // ---- point-cloud encoder: M = 4096 tokens, T = 2048, non-causal ----
    k_embed_pc<<<dim3(4096, 4), 256, 0, stream>>>(x, pos_w, pos_b, h);
    run_stack(h, 4096, 2048, 4, 0, pc_b1, pc_b2, pc_bo, pc_bqkv,
              pc_ln1b, pc_ln1g, pc_ln2b, pc_ln2g, pc_w1, pc_w2, pc_wo, pc_wqkv);
    k_ln<<<4096, 256, 0, stream>>>(h, pc_ng, pc_nb, tmpf);
    k_rowdot<<<4096, 256, 0, stream>>>(tmpf, pred_w, pred_b, logits);
    k_topk<<<2, 256, 0, stream>>>(logits, sel);

    // ---- AR decoder: gather + embed, M = 2046 tokens, T = 1023, causal ----
    k_embed_ar<<<dim3(1023, 4, 2), 256, 0, stream>>>(x, sel, ar_pos_w, ar_pos_b,
                                                     ar_val_w, ar_val_b, h);
    run_stack(h, 2046, 1023, 24, 1, ar_b1, ar_b2, ar_bo, ar_bqkv,
              ar_ln1b, ar_ln1g, ar_ln2b, ar_ln2g, ar_w1, ar_w2, ar_wo, ar_wqkv);
    k_ln<<<2046, 256, 0, stream>>>(h, ar_ng, ar_nb, tmpf);
    k_rowdot<<<2046, 256, 0, stream>>>(tmpf, head_w, head_b, (float*)d_out);
    k_out2<<<8, 256, 0, stream>>>(x, sel, (float*)d_out + 2046);
}
